// Attention_86517821211073
// MI455X (gfx1250) — matrix operand order repaired and checked by reading, not compiled, not run on hardware
//
#include <hip/hip_runtime.h>
#include <hip/hip_bf16.h>

// ---------------------------------------------------------------------------
// Attention block for MI455X (gfx1250, wave32, WMMA f32_16x16x32_f16).
// B=2, N=2048, DIM=1024, H=16, DH=64, INNER=1024.
// ---------------------------------------------------------------------------

typedef __attribute__((ext_vector_type(16))) _Float16 v16h;
typedef __attribute__((ext_vector_type(8)))  _Float16 v8h;
typedef __attribute__((ext_vector_type(8)))  float    v8f;

#define B_  2
#define N_  2048
#define DIM_ 1024
#define H_  16
#define DH_ 64
#define INNER_ 1024
#define NPAD_ 2080          // 65 key tiles of 32 covering N+1 = 2049 keys
#define EPS_ 1e-5f
#define NEGF_ (-3.0e38f)

__device__ __forceinline__ v16h mk16(v8h lo, v8h hi) {
    v16h r;
#pragma unroll
    for (int i = 0; i < 8; ++i) { r[i] = lo[i]; r[i + 8] = hi[i]; }
    return r;
}

__device__ __forceinline__ v8f wmma_f16(v16h a, v16h b, v8f c) {
    return __builtin_amdgcn_wmma_f32_16x16x32_f16(false, a, false, b,
                                                  (short)0, c, false, false);
}

// ---------------------------------------------------------------------------
// LayerNorm over rows of length 1024. 256 threads, 4 elems/thread.
// ---------------------------------------------------------------------------
template <typename OutT>
__global__ __launch_bounds__(256) void ln_kernel(const float* __restrict__ x,
                                                 const float* __restrict__ g,
                                                 OutT* __restrict__ out) {
    __shared__ float red[16];
    const int row = blockIdx.x;
    const int tid = threadIdx.x;
    const float4 v = *(const float4*)(x + (size_t)row * DIM_ + tid * 4);
    float s  = v.x + v.y + v.z + v.w;
    float ss = v.x * v.x + v.y * v.y + v.z * v.z + v.w * v.w;
#pragma unroll
    for (int d = 16; d; d >>= 1) {
        s  += __shfl_xor(s, d, 32);
        ss += __shfl_xor(ss, d, 32);
    }
    const int wid = tid >> 5, lane = tid & 31;
    if (lane == 0) { red[wid] = s; red[8 + wid] = ss; }
    __syncthreads();
    float ts = 0.f, tss = 0.f;
#pragma unroll
    for (int i = 0; i < 8; ++i) { ts += red[i]; tss += red[8 + i]; }
    const float mean = ts * (1.0f / DIM_);
    const float var  = tss * (1.0f / DIM_) - mean * mean;
    const float inv  = rsqrtf(var + EPS_);
    const float4 gv = *(const float4*)(g + tid * 4);
    OutT* o = out + (size_t)row * DIM_ + tid * 4;
    o[0] = (OutT)((v.x - mean) * inv * gv.x);
    o[1] = (OutT)((v.y - mean) * inv * gv.y);
    o[2] = (OutT)((v.z - mean) * inv * gv.z);
    o[3] = (OutT)((v.w - mean) * inv * gv.w);
}

// ---------------------------------------------------------------------------
// f32 -> f16 conversion.
// ---------------------------------------------------------------------------
__global__ __launch_bounds__(256) void cvt_kernel(const float* __restrict__ in,
                                                  _Float16* __restrict__ out, int n) {
    int i = blockIdx.x * 256 + threadIdx.x;
    if (i < n) out[i] = (_Float16)in[i];
}

// ---------------------------------------------------------------------------
// Tiled WMMA GEMM: C[M,Nc] = alpha * A[M,K] @ W[K,Nc], f16 inputs.
// Block tile 64x64 (8 waves as 4M x 2N, each wave a 16x32 strip -> A-fragment
// reused across two WMMAs per K-step). A staged row-major in LDS; W staged
// transposed so B-fragments are two contiguous ds_load_b128 per lane.
// ---------------------------------------------------------------------------
template <bool HALF_OUT>
__global__ __launch_bounds__(256) void gemm_kernel(const _Float16* __restrict__ A,
                                                   const _Float16* __restrict__ W,
                                                   void* __restrict__ Cout,
                                                   int M, int Nc, int K, float alpha) {
    __shared__ _Float16 sA[64 * 32];
    __shared__ _Float16 sBt[64 * 32];   // [col 0..63][k 0..31]
    const int tid = threadIdx.x;
    const int wid = tid >> 5, lane = tid & 31;
    const int l16 = lane & 15, g = lane >> 4;
    const int m0 = blockIdx.x * 64, n0 = blockIdx.y * 64;
    const int mt = wid & 3, nt = wid >> 2;     // nt = 0..1 -> cols nt*32 .. +31
    v8f acc0 = {}, acc1 = {};
    for (int k0 = 0; k0 < K; k0 += 32) {
        __syncthreads();
        {   // stage A: 64x32 halves, v8h per thread
            const int row = tid >> 2, ch = (tid & 3) * 8;
            *(v8h*)&sA[row * 32 + ch] =
                *(const v8h*)(A + (size_t)(m0 + row) * K + k0 + ch);
        }
        {   // stage W transposed: 32x64 region -> [col][k], v8h per thread
            const int r = tid >> 3, c8 = (tid & 7) * 8;
            v8h w = *(const v8h*)(W + (size_t)(k0 + r) * Nc + n0 + c8);
#pragma unroll
            for (int u = 0; u < 8; ++u) sBt[(c8 + u) * 32 + r] = w[u];
        }
        __syncthreads();
        const _Float16* ar = &sA[(mt * 16 + l16) * 32];
        v16h a = mk16(*(const v8h*)(ar + 8 * g), *(const v8h*)(ar + 16 + 8 * g));
        const _Float16* br0 = &sBt[(nt * 32 + l16) * 32];
        v16h b0 = mk16(*(const v8h*)(br0 + 8 * g), *(const v8h*)(br0 + 16 + 8 * g));
        const _Float16* br1 = &sBt[(nt * 32 + 16 + l16) * 32];
        v16h b1 = mk16(*(const v8h*)(br1 + 8 * g), *(const v8h*)(br1 + 16 + 8 * g));
        acc0 = wmma_f16(a, b0, acc0);
        acc1 = wmma_f16(a, b1, acc1);
    }
    const int col0 = n0 + nt * 32 + l16;
#pragma unroll
    for (int r = 0; r < 8; ++r) {
        const int row = m0 + mt * 16 + r + 8 * g;
        const float v0 = acc0[r] * alpha;
        const float v1 = acc1[r] * alpha;
        if (HALF_OUT) {
            ((_Float16*)Cout)[(size_t)row * Nc + col0]      = (_Float16)v0;
            ((_Float16*)Cout)[(size_t)row * Nc + col0 + 16] = (_Float16)v1;
        } else {
            ((float*)Cout)[(size_t)row * Nc + col0]      = v0;
            ((float*)Cout)[(size_t)row * Nc + col0 + 16] = v1;
        }
    }
}

// ---------------------------------------------------------------------------
// Build padded K/V: row 0 = null_kv, rows 1..N = kv GEMM output, rest zero.
// ---------------------------------------------------------------------------
__global__ __launch_bounds__(256) void padkv_kernel(const _Float16* __restrict__ kv,
                                                    const float* __restrict__ nullkv,
                                                    _Float16* __restrict__ kpad,
                                                    _Float16* __restrict__ vpad) {
    const int idx = blockIdx.x * 256 + threadIdx.x;
    const int total = B_ * NPAD_ * DH_;
    if (idx >= total) return;
    const int d = idx & 63;
    const int rr = idx >> 6;
    const int b = rr / NPAD_;
    const int row = rr % NPAD_;
    _Float16 kv_, vv_;
    if (row == 0) {
        kv_ = (_Float16)nullkv[d];
        vv_ = (_Float16)nullkv[DH_ + d];
    } else if (row <= N_) {
        const _Float16* src = kv + (size_t)(b * N_ + row - 1) * (2 * DH_);
        kv_ = src[d];
        vv_ = src[DH_ + d];
    } else {
        kv_ = (_Float16)0.f;
        vv_ = (_Float16)0.f;
    }
    kpad[idx] = kv_;
    vpad[idx] = vv_;
}

// ---------------------------------------------------------------------------
// Flash attention. Block = 8 waves x 16 query rows = 128 rows of one (b,h).
// Key tiles of 32 so P (16x32) is exactly one WMMA A operand.
// Bias/mask/causal applied branchlessly (clamped-index loads + selects).
// ---------------------------------------------------------------------------
__global__ __launch_bounds__(256) void attn_kernel(const _Float16* __restrict__ q,
                                                   const _Float16* __restrict__ kpad,
                                                   const _Float16* __restrict__ vpad,
                                                   const float* __restrict__ bias,
                                                   const float* __restrict__ nbias,
                                                   const unsigned char* __restrict__ mask,
                                                   _Float16* __restrict__ ctx) {
    __shared__ _Float16 sK[32 * 64];        // [j][d] row-major
    __shared__ _Float16 sVt[64 * 32];       // [d][j] transposed
    __shared__ _Float16 sP[8][16 * 32];     // per-wave P tile [i][j]

    const int tid = threadIdx.x;
    const int wid = tid >> 5, lane = tid & 31;
    const int l16 = lane & 15, g = lane >> 4;
    const int bx = blockIdx.x;
    const int b = blockIdx.y / H_;
    const int h = blockIdx.y % H_;
    const int i0w = bx * 128 + wid * 16;

    // Q fragments for this wave's 16 rows (q already scaled by dh^-0.5).
    const _Float16* qrow = q + (size_t)(b * N_ + i0w + l16) * INNER_ + h * DH_;
    const v16h aq0 = mk16(*(const v8h*)(qrow + 8 * g),      *(const v8h*)(qrow + 16 + 8 * g));
    const v16h aq1 = mk16(*(const v8h*)(qrow + 32 + 8 * g), *(const v8h*)(qrow + 48 + 8 * g));

    float m[8], l[8];
    v8f o0 = {}, o1 = {}, o2 = {}, o3 = {};
#pragma unroll
    for (int r = 0; r < 8; ++r) { m[r] = -__builtin_inff(); l[r] = 0.f; }

    const int nT = (bx * 4 + 5) < 65 ? (bx * 4 + 5) : 65;   // causal bound for block
    const _Float16* kb = kpad + (size_t)b * NPAD_ * DH_;
    const _Float16* vb = vpad + (size_t)b * NPAD_ * DH_;
    const float* biasH = bias + (size_t)h * N_ * N_;
    const unsigned char* maskb = mask + b * N_;
    const float nb = nbias[h];

    for (int t = 0; t < nT; ++t) {
        const int j0 = t * 32;
        __syncthreads();   // protect LDS from previous iteration's readers
        {   // stage K row-major, V transposed (32 rows x 64 halves each)
            const int jr = tid >> 3;
            const int d0 = (tid & 7) * 8;
            const size_t off = (size_t)(j0 + jr) * DH_ + d0;
            *(v8h*)&sK[jr * 64 + d0] = *(const v8h*)(kb + off);
            v8h vv = *(const v8h*)(vb + off);
#pragma unroll
            for (int u = 0; u < 8; ++u) sVt[(d0 + u) * 32 + jr] = vv[u];
            if (t + 1 < nT) {   // prefetch next K/V tile
                __builtin_prefetch(kb + off + 32 * DH_, 0, 0);
                __builtin_prefetch(vb + off + 32 * DH_, 0, 0);
            }
        }
        __syncthreads();

        const bool active = (j0 <= i0w + 16);   // wave-uniform
        if (active) {
            // S = q @ k^T, columns split in two 16-wide halves.
            v8f c0 = {}, c1 = {};
            {
                const _Float16* kr = &sK[l16 * 64];
                v16h b0 = mk16(*(const v8h*)(kr + 8 * g), *(const v8h*)(kr + 16 + 8 * g));
                v16h b1 = mk16(*(const v8h*)(kr + 32 + 8 * g), *(const v8h*)(kr + 48 + 8 * g));
                c0 = wmma_f16(aq0, b0, c0);
                c0 = wmma_f16(aq1, b1, c0);
            }
            {
                const _Float16* kr = &sK[(16 + l16) * 64];
                v16h b0 = mk16(*(const v8h*)(kr + 8 * g), *(const v8h*)(kr + 16 + 8 * g));
                v16h b1 = mk16(*(const v8h*)(kr + 32 + 8 * g), *(const v8h*)(kr + 48 + 8 * g));
                c1 = wmma_f16(aq0, b0, c1);
                c1 = wmma_f16(aq1, b1, c1);
            }
            // ---- branchless bias + padding mask + causal + online softmax ----
            const int jj0 = j0 + l16;
            const int jj1 = jj0 + 16;
            // clamped bias column (valid even when jj==0 or jj>N)
            const int jc0 = (jj0 < 1 ? 0 : (jj0 > N_ ? N_ - 1 : jj0 - 1));
            const int jc1 = (jj1 > N_ ? N_ - 1 : jj1 - 1);
            const unsigned char mv0 = maskb[jc0];      // hoisted: column-only
            const unsigned char mv1 = maskb[jc1];
            const bool key0 = (jj0 == 0);
            const bool ok0 = key0 | ((jj0 <= N_) & (mv0 != 0));
            const bool ok1 = (jj1 <= N_) & (mv1 != 0);
            const float* bc0 = biasH + jc0;
            const float* bc1 = biasH + jc1;
#pragma unroll
            for (int r = 0; r < 8; ++r) {
                const int i = i0w + r + 8 * g;
                float bv0 = bc0[(size_t)i * N_];       // unconditional, coalesced
                float bv1 = bc1[(size_t)i * N_];
                bv0 = key0 ? nb : bv0;
                float s0 = c0[r] + bv0;
                float s1 = c1[r] + bv1;
                s0 = (ok0 & (jj0 <= i + 1)) ? s0 : NEGF_;
                s1 = (ok1 & (jj1 <= i + 1)) ? s1 : NEGF_;
                float mx = fmaxf(s0, s1);
#pragma unroll
                for (int d = 8; d; d >>= 1) mx = fmaxf(mx, __shfl_xor(mx, d, 32));
                const float nm = fmaxf(m[r], mx);
                const float f = __expf(m[r] - nm);
                const float e0 = __expf(s0 - nm);
                const float e1 = __expf(s1 - nm);
                float rs = e0 + e1;
#pragma unroll
                for (int d = 8; d; d >>= 1) rs += __shfl_xor(rs, d, 32);
                l[r] = l[r] * f + rs;
                m[r] = nm;
                o0[r] *= f; o1[r] *= f; o2[r] *= f; o3[r] *= f;
                sP[wid][(r + 8 * g) * 32 + l16]      = (_Float16)e0;
                sP[wid][(r + 8 * g) * 32 + 16 + l16] = (_Float16)e1;
            }
        }
        __syncthreads();   // P visible; LDS settled block-wide
        if (active) {
            const _Float16* pr = &sP[wid][l16 * 32];
            const v16h ap = mk16(*(const v8h*)(pr + 8 * g), *(const v8h*)(pr + 16 + 8 * g));
#pragma unroll
            for (int fr = 0; fr < 4; ++fr) {
                const _Float16* vr = &sVt[(fr * 16 + l16) * 32];
                v16h bv = mk16(*(const v8h*)(vr + 8 * g), *(const v8h*)(vr + 16 + 8 * g));
                v8f* op = (fr == 0) ? &o0 : (fr == 1) ? &o1 : (fr == 2) ? &o2 : &o3;
                *op = wmma_f16(ap, bv, *op);
            }
        }
    }

    // normalize and write context [B*N, INNER]
    _Float16* cbase = ctx + (size_t)(b * N_ + i0w) * INNER_ + h * DH_;
#pragma unroll
    for (int r = 0; r < 8; ++r) {
        const float inv = (l[r] > 0.f) ? (1.f / l[r]) : 0.f;
        const size_t roff = (size_t)(r + 8 * g) * INNER_;
        cbase[roff + 0 * 16 + l16] = (_Float16)(o0[r] * inv);
        cbase[roff + 1 * 16 + l16] = (_Float16)(o1[r] * inv);
        cbase[roff + 2 * 16 + l16] = (_Float16)(o2[r] * inv);
        cbase[roff + 3 * 16 + l16] = (_Float16)(o3[r] * inv);
    }
}

// ---------------------------------------------------------------------------
// Launcher
// ---------------------------------------------------------------------------
extern "C" void kernel_launch(void* const* d_in, const int* in_sizes, int n_in,
                              void* d_out, int out_size, void* d_ws, size_t ws_size,
                              hipStream_t stream) {
    const float* x      = (const float*)d_in[0];
    const float* abias  = (const float*)d_in[1];
    const unsigned char* mask = (const unsigned char*)d_in[2];
    const float* g_in   = (const float*)d_in[3];
    const float* nbias  = (const float*)d_in[4];
    const float* nullkv = (const float*)d_in[5];
    const float* Wq     = (const float*)d_in[6];
    const float* Wkv    = (const float*)d_in[7];
    const float* Wo     = (const float*)d_in[8];
    const float* g_out  = (const float*)d_in[9];
    float* out = (float*)d_out;

    char* w = (char*)d_ws;
    _Float16* xn   = (_Float16*)w;  w += (size_t)B_ * N_ * DIM_ * 2;      // 8 MB
    _Float16* Wqh  = (_Float16*)w;  w += (size_t)DIM_ * INNER_ * 2;       // 2 MB
    _Float16* Wkvh = (_Float16*)w;  w += (size_t)DIM_ * 2 * DH_ * 2;      // 256 KB
    _Float16* Woh  = (_Float16*)w;  w += (size_t)INNER_ * DIM_ * 2;       // 2 MB
    _Float16* qh   = (_Float16*)w;  w += (size_t)B_ * N_ * INNER_ * 2;    // 8 MB
    _Float16* kvh  = (_Float16*)w;  w += (size_t)B_ * N_ * 2 * DH_ * 2;   // 1 MB
    _Float16* kpad = (_Float16*)w;  w += (size_t)B_ * NPAD_ * DH_ * 2;    // 520 KB
    _Float16* vpad = (_Float16*)w;  w += (size_t)B_ * NPAD_ * DH_ * 2;    // 520 KB
    _Float16* ctx  = (_Float16*)w;  w += (size_t)B_ * N_ * INNER_ * 2;    // 8 MB
    float*    proj = (float*)w;     w += (size_t)B_ * N_ * DIM_ * 4;      // 16 MB

    const int rows = B_ * N_;                       // 4096

    // 1) input layernorm -> f16
    ln_kernel<_Float16><<<rows, 256, 0, stream>>>(x, g_in, xn);

    // 2) weight conversion to f16
    cvt_kernel<<<(DIM_ * INNER_ + 255) / 256, 256, 0, stream>>>(Wq, Wqh, DIM_ * INNER_);
    cvt_kernel<<<(DIM_ * 2 * DH_ + 255) / 256, 256, 0, stream>>>(Wkv, Wkvh, DIM_ * 2 * DH_);
    cvt_kernel<<<(INNER_ * DIM_ + 255) / 256, 256, 0, stream>>>(Wo, Woh, INNER_ * DIM_);

    // 3) q = (xn @ Wq) * dh^-0.5 ; kv = xn @ Wkv
    gemm_kernel<true><<<dim3(rows / 64, INNER_ / 64), 256, 0, stream>>>(
        xn, Wqh, qh, rows, INNER_, DIM_, 0.125f);
    gemm_kernel<true><<<dim3(rows / 64, (2 * DH_) / 64), 256, 0, stream>>>(
        xn, Wkvh, kvh, rows, 2 * DH_, DIM_, 1.0f);

    // 4) pad k/v with null row
    padkv_kernel<<<(B_ * NPAD_ * DH_ + 255) / 256, 256, 0, stream>>>(kvh, nullkv, kpad, vpad);

    // 5) flash attention
    attn_kernel<<<dim3(N_ / 128, B_ * H_), 256, 0, stream>>>(
        qh, kpad, vpad, abias, nbias, mask, ctx);

    // 6) output projection (f32 accum out)
    gemm_kernel<false><<<dim3(rows / 64, DIM_ / 64), 256, 0, stream>>>(
        ctx, Woh, proj, rows, DIM_, INNER_, 1.0f);

    // 7) final layernorm -> f32 output
    ln_kernel<float><<<rows, 256, 0, stream>>>(proj, g_out, out);
}
